// SphericalHarmonicsConvolution_51969104282137
// MI455X (gfx1250) — matrix operand order repaired and checked by reading, not compiled, not run on hardware
//
#include <hip/hip_runtime.h>
#include <hip/hip_bf16.h>

#define NN 50000
#define EE 500000
#define ALPHA 0.20412414523193154f   /* 1/sqrt(24) */
#define C110f 0.5773502691896258f    /* 1/sqrt(3) */
#define EPSf 1e-5f

// LDS/weight-image layout (bytes):
//  [0,      73728)  w3cm  f16 [576][64] column-major
//  [73728,  81920)  w2cm  f16 [64][64]  column-major
//  [81920,  86016)  w1cm  f16 [64][32]  column-major, K 16..31 zero-padded
//  [86016,  86272)  b1s   f32 [64]
//  [86272,  86528)  b2s   f32 [64]
//  [86528,  88832)  b3s   f32 [576]
#define WIMG_BYTES 88832

typedef __attribute__((ext_vector_type(16))) _Float16 v16h;
typedef __attribute__((ext_vector_type(8)))  _Float16 v8h;
typedef __attribute__((ext_vector_type(8)))  float    v8f;

union U16 { v16h v; v8h h[2]; };

// CDNA5 16-bit A fragment (16x32, row M = lane&15):
//  elems 0..7  : K = kb + half*8 + (0..7)
//  elems 8..15 : K = kb + 16 + half*8 + (0..7)
__device__ __forceinline__ v16h fragA(const _Float16* row, int kb, int half) {
    U16 u;
    u.h[0] = *(const v8h*)(row + kb + half * 8);
    u.h[1] = *(const v8h*)(row + kb + 16 + half * 8);
    return u.v;
}
// CDNA5 16-bit B fragment (32x16, col N = lane&15), same K order as fragA:
//  elems 0..7 : K = kb + half*8 + (0..7); elems 8..15 : K = kb + 16 + half*8 + (0..7)
__device__ __forceinline__ v16h fragB(const _Float16* col, int kb, int half) {
    U16 u;
    u.h[0] = *(const v8h*)(col + kb + half * 8);
    u.h[1] = *(const v8h*)(col + kb + 16 + half * 8);
    return u.v;
}

__device__ __forceinline__ v8f wmma16(v16h a, v16h b, v8f c) {
    return __builtin_amdgcn_wmma_f32_16x16x32_f16(false, a, false, b, (short)0, c, false, false);
}

__device__ __forceinline__ v8f splat8(float x) {
    v8f v;
#pragma unroll
    for (int j = 0; j < 8; ++j) v[j] = x;
    return v;
}

// Fast SiLU: v_exp_f32 + v_rcp_f32 (no IEEE divide sequence)
__device__ __forceinline__ float silu(float x) {
    return x * __builtin_amdgcn_rcpf(1.f + __expf(-x));
}

__device__ __forceinline__ float comp4(const float4& v, int q) {
    return ((const float*)&v)[q];   // q is compile-time under full unroll
}

#define LDS_SYNC() asm volatile("s_wait_dscnt 0" ::: "memory")

// ---------------- Kernel 0: zero workspace ----------------
__global__ void zero_kernel(float* __restrict__ p, int n) {
    int i = blockIdx.x * 256 + threadIdx.x;
    if (i < n) p[i] = 0.f;
}

// ---------------- Kernel 0b: build the shared-weight LDS image once ----------------
__global__ void prep_kernel(
    const float* __restrict__ w1, const float* __restrict__ b1,
    const float* __restrict__ w2, const float* __restrict__ b2,
    const float* __restrict__ w3, const float* __restrict__ b3,
    char* __restrict__ img)
{
    int i = blockIdx.x * 256 + threadIdx.x;
    _Float16* hp = (_Float16*)img;
    if (i < 36864) {                       // w3cm
        int n = i >> 6, k = i & 63;
        hp[i] = (_Float16)w3[k * 576 + n];
    } else if (i < 40960) {                // w2cm
        int j = i - 36864;
        int n = j >> 6, k = j & 63;
        hp[i] = (_Float16)w2[k * 64 + n];
    } else if (i < 43008) {                // w1cm (zero-padded K)
        int j = i - 40960;
        int n = j >> 5, k = j & 31;
        hp[i] = (k < 16) ? (_Float16)w1[k * 64 + n] : (_Float16)0.f;
    } else if (i < 43712) {                // biases (f32)
        int j = i - 43008;
        float* fp = (float*)(img + 86016);
        fp[j] = (j < 64) ? b1[j] : (j < 128) ? b2[j - 64] : b3[j - 128];
    }
}

// ---------------- Kernel 1: fused edge MLP (WMMA) + tensor product + scatter ----------------
// 8 waves/block, each wave owns a 16-edge tile.
__global__ void __launch_bounds__(256) edge_kernel(
    const float* __restrict__ h, const int* __restrict__ ei,
    const float* __restrict__ esh, const float* __restrict__ ef,
    const char* __restrict__ wimg,
    float* __restrict__ agg, float* __restrict__ cnt)
{
    extern __shared__ char smem[];
    _Float16* w3cm = (_Float16*)(smem);
    _Float16* w2cm = (_Float16*)(smem + 73728);
    _Float16* w1cm = (_Float16*)(smem + 81920);
    float*    b1s  = (float*)(smem + 86016);
    float*    b2s  = b1s + 64;
    float*    b3s  = b2s + 64;
    char*     wbase = smem + WIMG_BYTES;               // per-wave regions, 7488 B each
    (void)w2cm; (void)w1cm; (void)b1s; (void)b2s;

    const int tid = threadIdx.x;

    // Bulk-copy the prebuilt weight image (L2-resident) into LDS, b128 granularity.
    {
        const float4* src = (const float4*)wimg;
        float4* dst = (float4*)smem;
        for (int idx = tid; idx < WIMG_BYTES / 16; idx += 256) dst[idx] = src[idx];
    }
    __syncthreads();

    const int wave = tid >> 5, lane = tid & 31;
    const int laneN = lane & 15, half = lane >> 4;
    const int base = (blockIdx.x * 8 + wave) * 16;
    if (base >= EE) return;                             // E is a multiple of 16: no partial tiles

    char* wp = wbase + wave * 7488;
    _Float16* hidA = (_Float16*)(wp);                   // [16][64] f16    2048 B
    _Float16* hidB = (_Float16*)(wp + 2048);            // [16][64] f16    2048 B
    float*    hrow = (float*)(wp + 4096);               // [16][40] f32    2560 B
    float*    shs  = (float*)(wp + 6656);               // [16][4]          256 B
    float*    vdot = (float*)(wp + 6912);               // [16][8]          512 B
    int*      dsts = (int*)(wp + 7424);                 // [16]              64 B

    // ---- gather per-edge data ----
    if (lane < 16) {
        int e = base + lane;
        dsts[lane] = ei[EE + e];
        *(float4*)(shs + lane * 4) = *(const float4*)(esh + (size_t)e * 4);
    }
    {
        int e = base + laneN;
        int src = ei[e];
        const float4* hp = (const float4*)(h + (size_t)src * 40 + half * 20);
        float4* dq = (float4*)(hrow + laneN * 40 + half * 20);
#pragma unroll
        for (int t = 0; t < 5; ++t) dq[t] = hp[t];
    }
    LDS_SYNC();
    // vdot[M][i] = v[M][i] . sh1[M]
#pragma unroll
    for (int r = 0; r < 4; ++r) {
        int idx = lane + r * 32;
        int M = idx >> 3, i = idx & 7;
        const float* vv = hrow + M * 40 + 16 + i * 3;
        const float* sp = shs + M * 4;
        vdot[idx] = vv[0] * sp[1] + vv[1] * sp[2] + vv[2] * sp[3];
    }

    // Loop-invariant per-lane base pointers (imm-offset addressing below).
    const _Float16* cb1 = w1cm + laneN * 32;            // B col base, layer 1
    const _Float16* cb2 = w2cm + laneN * 64;            // B col base, layer 2
    const _Float16* cb3 = w3cm + laneN * 64;            // B col base, layer 3
    const float*    bb1 = b1s + laneN;
    const float*    bb2 = b2s + laneN;
    const float*    bb3 = b3s + laneN;
    _Float16*       stA = hidA + half * 512 + laneN;    // store base: + j*64 + nt*16
    _Float16*       stB = hidB + half * 512 + laneN;
    const float*    hr  = hrow + half * 320;            // + j*40 + i
    const float*    vd  = vdot + half * 64;             // + j*8 + t
    const float*    ss  = shs + half * 32;              // + j*4 + c
    const int*      dsb = dsts + half * 8;              // + j

    // ---- layer 1: (16x16) @ (16x64), K padded to 32; bias preloaded in C ----
    v16h a1;
    {
        int e = base + laneN;
        const float* fp = ef + (size_t)e * 16 + half * 8;
        U16 u; v8h t, z;
#pragma unroll
        for (int k = 0; k < 8; ++k) { t[k] = (_Float16)fp[k]; z[k] = (_Float16)0.f; }
        u.h[0] = t; u.h[1] = z;
        a1 = u.v;
    }
#pragma unroll
    for (int nt = 0; nt < 4; ++nt) {
        v8f c = splat8(bb1[nt * 16]);
        c = wmma16(a1, fragB(cb1 + nt * 512, 0, half), c);
#pragma unroll
        for (int j = 0; j < 8; ++j)
            stA[j * 64 + nt * 16] = (_Float16)silu(c[j]);
    }
    LDS_SYNC();

    // ---- layer 2: (16x64) @ (64x64) ----
    {
        const _Float16* ar = hidA + laneN * 64;
        v16h a0 = fragA(ar, 0, half);
        v16h a1b = fragA(ar, 32, half);
#pragma unroll
        for (int nt = 0; nt < 4; ++nt) {
            v8f c = splat8(bb2[nt * 16]);
            c = wmma16(a0, fragB(cb2 + nt * 1024, 0, half), c);
            c = wmma16(a1b, fragB(cb2 + nt * 1024, 32, half), c);
#pragma unroll
            for (int j = 0; j < 8; ++j)
                stB[j * 64 + nt * 16] = (_Float16)silu(c[j]);
        }
    }
    LDS_SYNC();

    // ---- layer 3: (16x64) @ (64x576), consume D tiles in-register ----
    const _Float16* ar = hidB + laneN * 64;
    v16h a0 = fragA(ar, 0, half);
    v16h a1b = fragA(ar, 32, half);

    v8f accS1 = {}, accS4 = {}, accV2 = {}, accV3x = {}, accV3y = {}, accV3z = {};

    // w1 block: cols [0,256), col = i*16+o  => o = laneN, i = nt
    // s operands fetched 4-at-a-time as ds_load_b128.
#pragma unroll
    for (int g = 0; g < 4; ++g) {
        float4 sv[8];
#pragma unroll
        for (int j = 0; j < 8; ++j) sv[j] = *(const float4*)(hr + j * 40 + g * 4);
#pragma unroll
        for (int q = 0; q < 4; ++q) {
            int nt = g * 4 + q;
            v8f c = splat8(bb3[nt * 16]);
            c = wmma16(a0, fragB(cb3 + nt * 1024, 0, half), c);
            c = wmma16(a1b, fragB(cb3 + nt * 1024, 32, half), c);
            v8f sj;
#pragma unroll
            for (int j = 0; j < 8; ++j) sj[j] = comp4(sv[j], q);
            accS1 = __builtin_elementwise_fma(sj, c, accS1);
        }
    }
    // w2 block: cols [256,384), col = 256 + i*8+o => i = 2t+(laneN>>3), o = laneN&7
#pragma unroll
    for (int t = 0; t < 8; ++t) {
        int nt = 16 + t;
        v8f c = splat8(bb3[nt * 16]);
        c = wmma16(a0, fragB(cb3 + nt * 1024, 0, half), c);
        c = wmma16(a1b, fragB(cb3 + nt * 1024, 32, half), c);
        const float* sp = hr + 2 * t + (laneN >> 3);
        v8f sj;
#pragma unroll
        for (int j = 0; j < 8; ++j) sj[j] = sp[j * 40];
        accV2 = __builtin_elementwise_fma(sj, c, accV2);
    }
    // w3 block: cols [384,448), col = 384 + i*8+o
#pragma unroll
    for (int t = 0; t < 4; ++t) {
        int nt = 24 + t;
        v8f c = splat8(bb3[nt * 16]);
        c = wmma16(a0, fragB(cb3 + nt * 1024, 0, half), c);
        c = wmma16(a1b, fragB(cb3 + nt * 1024, 32, half), c);
        int i = 2 * t + (laneN >> 3);
        const float* vp = hr + 16 + i * 3;
        v8f vx, vy, vz;
#pragma unroll
        for (int j = 0; j < 8; ++j) {
            vx[j] = vp[j * 40 + 0];
            vy[j] = vp[j * 40 + 1];
            vz[j] = vp[j * 40 + 2];
        }
        accV3x = __builtin_elementwise_fma(vx, c, accV3x);
        accV3y = __builtin_elementwise_fma(vy, c, accV3y);
        accV3z = __builtin_elementwise_fma(vz, c, accV3z);
    }
    // w4 block: cols [448,576), col = 448 + i*16+o => o = laneN, i = t
    // vdot operands fetched 4-at-a-time as ds_load_b128.
#pragma unroll
    for (int g = 0; g < 2; ++g) {
        float4 vq[8];
#pragma unroll
        for (int j = 0; j < 8; ++j) vq[j] = *(const float4*)(vd + j * 8 + g * 4);
#pragma unroll
        for (int q = 0; q < 4; ++q) {
            int nt = 28 + g * 4 + q;
            v8f c = splat8(bb3[nt * 16]);
            c = wmma16(a0, fragB(cb3 + nt * 1024, 0, half), c);
            c = wmma16(a1b, fragB(cb3 + nt * 1024, 32, half), c);
            v8f sj;
#pragma unroll
            for (int j = 0; j < 8; ++j) sj[j] = comp4(vq[j], q);
            accS4 = __builtin_elementwise_fma(sj, c, accS4);
        }
    }

    // ---- emit: scatter messages into agg via global f32 atomics ----
#pragma unroll
    for (int j = 0; j < 8; ++j) {
        float val = ALPHA * (ss[j * 4] * accS1[j] + C110f * accS4[j]);
        atomicAdd(agg + (size_t)dsb[j] * 40 + laneN, val);
    }
#pragma unroll
    for (int j = 0; j < 8; ++j) {
        float s2  = accV2[j]  + __shfl_xor(accV2[j], 8, 32);
        float s30 = accV3x[j] + __shfl_xor(accV3x[j], 8, 32);
        float s31 = accV3y[j] + __shfl_xor(accV3y[j], 8, 32);
        float s32 = accV3z[j] + __shfl_xor(accV3z[j], 8, 32);
        if ((lane & 8) == 0) {
            int o = laneN & 7;
            float sh0 = ss[j * 4];
            float* p = agg + (size_t)dsb[j] * 40 + 16 + o * 3;
            const float k = ALPHA * C110f;
            atomicAdd(p + 0, k * (ss[j * 4 + 1] * s2 + sh0 * s30));
            atomicAdd(p + 1, k * (ss[j * 4 + 2] * s2 + sh0 * s31));
            atomicAdd(p + 2, k * (ss[j * 4 + 3] * s2 + sh0 * s32));
        }
    }
    if (lane < 16) atomicAdd(cnt + dsts[lane], 1.0f);
}

// ---------------- Kernel 2: per-node self-interaction + pre-norm + stat partials ----------------
__global__ void __launch_bounds__(256) node_kernel(
    const float* __restrict__ h, const float* __restrict__ si_w0,
    const float* __restrict__ si_w1, const float* __restrict__ agg,
    const float* __restrict__ cnt, float* __restrict__ pre, float* __restrict__ stats)
{
    __shared__ float sw0[256];
    __shared__ float sw1[64];
    __shared__ float part[40];
    int tid = threadIdx.x;
    sw0[tid] = si_w0[tid];
    if (tid < 64) sw1[tid] = si_w1[tid];
    if (tid < 40) part[tid] = 0.f;
    __syncthreads();

    int n = blockIdx.x * 256 + tid;
    if (n < NN) {
        const float* hp = h + (size_t)n * 40;
        const float* ap = agg + (size_t)n * 40;
        float hs[16], hv[24], ps[16], pv[24];
#pragma unroll
        for (int i = 0; i < 16; ++i) hs[i] = hp[i];
#pragma unroll
        for (int i = 0; i < 24; ++i) hv[i] = hp[16 + i];
        float inv = __builtin_amdgcn_rcpf(fmaxf(cnt[n], 1.f));
#pragma unroll
        for (int o = 0; o < 16; ++o) {
            float acc = 0.f;
#pragma unroll
            for (int i = 0; i < 16; ++i) acc += hs[i] * sw0[i * 16 + o];
            ps[o] = ap[o] * inv + acc * 0.25f;          // 1/sqrt(16)
        }
#pragma unroll
        for (int o = 0; o < 8; ++o)
#pragma unroll
            for (int m = 0; m < 3; ++m) {
                float acc = 0.f;
#pragma unroll
                for (int i = 0; i < 8; ++i) acc += hv[i * 3 + m] * sw1[i * 8 + o];
                pv[o * 3 + m] = ap[16 + o * 3 + m] * inv + acc * 0.35355339059327373f; // 1/sqrt(8)
            }
        float* pp = pre + (size_t)n * 40;
#pragma unroll
        for (int o = 0; o < 16; ++o) pp[o] = ps[o];
#pragma unroll
        for (int i = 0; i < 24; ++i) pp[16 + i] = pv[i];
#pragma unroll
        for (int o = 0; o < 16; ++o) {
            atomicAdd(&part[o], ps[o]);
            atomicAdd(&part[16 + o], ps[o] * ps[o]);
        }
#pragma unroll
        for (int o = 0; o < 8; ++o) {
            float q = pv[o * 3] * pv[o * 3] + pv[o * 3 + 1] * pv[o * 3 + 1] + pv[o * 3 + 2] * pv[o * 3 + 2];
            atomicAdd(&part[32 + o], q);
        }
    }
    __syncthreads();
    if (tid < 40) atomicAdd(&stats[tid], part[tid]);
}

// ---------------- Kernel 3: normalize + residual ----------------
__global__ void __launch_bounds__(256) final_kernel(
    const float* __restrict__ h, const float* __restrict__ pre,
    const float* __restrict__ stats, const float* __restrict__ gamma0,
    const float* __restrict__ beta0, const float* __restrict__ gamma1,
    float* __restrict__ out)
{
    __shared__ float sc[16], of[16], vg[8];
    int tid = threadIdx.x;
    if (tid < 16) {
        float mu = stats[tid] * (1.f / NN);
        float var = stats[16 + tid] * (1.f / NN) - mu * mu;
        float is = rsqrtf(var + EPSf) * gamma0[tid];
        sc[tid] = is;
        of[tid] = beta0[tid] - mu * is;
    }
    if (tid < 8) {
        float vn = stats[32 + tid] * (1.f / (3.f * NN));
        vg[tid] = rsqrtf(vn + EPSf) * gamma1[tid];
    }
    __syncthreads();
    int n = blockIdx.x * 256 + tid;
    if (n >= NN) return;
    const float* pp = pre + (size_t)n * 40;
    const float* hp = h + (size_t)n * 40;
    float* op = out + (size_t)n * 40;
#pragma unroll
    for (int o = 0; o < 16; ++o) op[o] = pp[o] * sc[o] + of[o] + hp[o];
#pragma unroll
    for (int o = 0; o < 8; ++o)
#pragma unroll
        for (int m = 0; m < 3; ++m) {
            int idx = 16 + o * 3 + m;
            op[idx] = pp[idx] * vg[o] + hp[idx];
        }
}

extern "C" void kernel_launch(void* const* d_in, const int* in_sizes, int n_in,
                              void* d_out, int out_size, void* d_ws, size_t ws_size,
                              hipStream_t stream) {
    (void)in_sizes; (void)n_in; (void)out_size; (void)ws_size;
    const float* h     = (const float*)d_in[0];
    const int*   ei    = (const int*)d_in[1];
    const float* esh   = (const float*)d_in[2];
    const float* ef    = (const float*)d_in[3];
    const float* w1    = (const float*)d_in[4];
    const float* b1    = (const float*)d_in[5];
    const float* w2    = (const float*)d_in[6];
    const float* b2    = (const float*)d_in[7];
    const float* w3    = (const float*)d_in[8];
    const float* b3    = (const float*)d_in[9];
    const float* si_w0 = (const float*)d_in[10];
    const float* si_w1 = (const float*)d_in[11];
    const float* g0    = (const float*)d_in[12];
    const float* be0   = (const float*)d_in[13];
    const float* g1    = (const float*)d_in[14];

    float* ws    = (float*)d_ws;
    float* agg   = ws;                               // N*40
    float* cnt   = ws + (size_t)NN * 40;             // N
    float* stats = ws + (size_t)NN * 41;             // 40
    float* pre   = ws + (size_t)NN * 41 + 40;        // N*40
    char*  wimg  = (char*)(ws + (size_t)NN * 81 + 40); // 88832 B weight image (16B aligned)

    int zn = NN * 41 + 40;
    zero_kernel<<<(zn + 255) / 256, 256, 0, stream>>>(ws, zn);
    prep_kernel<<<(43712 + 255) / 256, 256, 0, stream>>>(w1, b1, w2, b2, w3, b3, wimg);

    size_t shmem = WIMG_BYTES + 8 * 7488;            // 148736 bytes
    int nblk = (EE / 16 + 7) / 8;                    // 3907
    edge_kernel<<<nblk, 256, shmem, stream>>>(h, ei, esh, ef, wimg, agg, cnt);

    node_kernel<<<(NN + 255) / 256, 256, 0, stream>>>(h, si_w0, si_w1, agg, cnt, pre, stats);
    final_kernel<<<(NN + 255) / 256, 256, 0, stream>>>(h, pre, stats, g0, be0, g1, (float*)d_out);
}
